// KimiMoE_18365280157741
// MI455X (gfx1250) — matrix operand order repaired and checked by reading, not compiled, not run on hardware
//
#include <hip/hip_runtime.h>

typedef unsigned short u16;
typedef unsigned int   u32;
typedef __bf16 bf16_t;
typedef bf16_t v16bf __attribute__((ext_vector_type(16)));
typedef float  v8f   __attribute__((ext_vector_type(8)));

union Frag { v16bf v; uint4 q[2]; };

#define T_TOK  1024
#define H_DIM  2048
#define E_NUM  32
#define I_DIM  1024
#define IS_DIM 2048

__device__ __forceinline__ u16 f2bf(float f) {
  union { float f; u32 u; } v; v.f = f;
  u32 u = v.u;
  u32 r = (u + 0x7FFFu + ((u >> 16) & 1u)) >> 16;  // RNE
  return (u16)r;
}

__device__ __forceinline__ float silu_f(float g) {
  return g / (1.f + __expf(-g));
}

// LDS byte offset of a generic pointer to __shared__ (generic LDS addr low 32 bits)
__device__ __forceinline__ u32 lds_off(const void* p) {
  return (u32)(uintptr_t)p;
}

// CDNA5 async DMA: global -> LDS, 16B per lane, tracked by ASYNCcnt
__device__ __forceinline__ void async_b128(u32 lds_dst, const void* gsrc) {
  asm volatile("global_load_async_to_lds_b128 %0, %1, off"
               :: "v"(lds_dst), "v"((unsigned long long)(uintptr_t)gsrc)
               : "memory");
}
__device__ __forceinline__ void wait_async0() {
#if __has_builtin(__builtin_amdgcn_s_wait_asynccnt)
  __builtin_amdgcn_s_wait_asynccnt(0);
#else
  asm volatile("s_wait_asynccnt 0x0" ::: "memory");
#endif
}

// ---------------- 1) cast hidden_states fp32 -> bf16 ----------------
__global__ __launch_bounds__(256)
void cast_bf16_kernel(const float* __restrict__ src, u16* __restrict__ dst, int n4) {
  int i = blockIdx.x * 256 + threadIdx.x;
  if (i < n4) {
    float4 f = ((const float4*)src)[i];
    ushort4 r;
    r.x = f2bf(f.x); r.y = f2bf(f.y); r.z = f2bf(f.z); r.w = f2bf(f.w);
    ((ushort4*)dst)[i] = r;
  }
}

// ---------------- 2) routing: one wave32 per token, one expert per lane ----------------
__global__ __launch_bounds__(32)
void route_kernel(const float* __restrict__ x, const float* __restrict__ gw,
                  const float* __restrict__ bias, float* __restrict__ combine,
                  int* __restrict__ counts, int* __restrict__ tokIdx) {
  const int t = blockIdx.x;
  const int e = threadIdx.x;  // 0..31
  const float* __restrict__ xr = x + (size_t)t * H_DIM;
  float acc = 0.f;
  for (int h = 0; h < H_DIM; ++h) acc = fmaf(xr[h], gw[h * E_NUM + e], acc);
  float score = 1.f / (1.f + __expf(-acc));
  float sfc = score + bias[e];

  __shared__ float sS[E_NUM], sF[E_NUM];
  sS[e] = score; sF[e] = sfc;
  combine[(size_t)t * E_NUM + e] = 0.f;
  __syncthreads();

  if (e == 0) {
    float gs[4];
    for (int g = 0; g < 4; ++g) {
      float m1 = -1e30f, m2 = -1e30f;
      for (int j = 0; j < 8; ++j) {
        float v = sF[g * 8 + j];
        if (v > m1) { m2 = m1; m1 = v; } else if (v > m2) { m2 = v; }
      }
      gs[g] = m1 + m2;
    }
    int g1 = 0; for (int g = 1; g < 4; ++g) if (gs[g] > gs[g1]) g1 = g;
    int g2 = -1; for (int g = 0; g < 4; ++g) { if (g == g1) continue; if (g2 < 0 || gs[g] > gs[g2]) g2 = g; }
    for (int i = 0; i < E_NUM; ++i) { int g = i >> 3; if (g != g1 && g != g2) sF[i] = -1e30f; }
    int sel[8]; float wsel[8]; float wsum = 0.f;
    for (int k = 0; k < 8; ++k) {
      int bi = 0;
      for (int i = 1; i < E_NUM; ++i) if (sF[i] > sF[bi]) bi = i;
      sel[k] = bi; wsel[k] = sS[bi]; wsum += sS[bi]; sF[bi] = -1e30f;
    }
    float scl = 2.5f / wsum;  // renorm * routed_scaling_factor
    for (int k = 0; k < 8; ++k) {
      int ei = sel[k];
      combine[(size_t)t * E_NUM + ei] = wsel[k] * scl;
      int pos = atomicAdd(&counts[ei], 1);
      tokIdx[ei * T_TOK + pos] = t;
    }
  }
}

// ---------------- 3) exclusive scan of 16-padded counts ----------------
__global__ void scan_kernel(const int* __restrict__ counts, int* __restrict__ offsets) {
  if (threadIdx.x == 0) {
    int acc = 0;
    for (int e = 0; e < E_NUM; ++e) { offsets[e] = acc; acc += (counts[e] + 15) & ~15; }
    offsets[E_NUM] = acc;
  }
}

// ---------------- 4) fused gate/up GEMM + SiLU (shared & routed) ----------------
// block: 256 thr = 8 waves; tile BM=128 (16 rows/wave), BN=64 (4 n-subtiles/wave), BK=32
__global__ __launch_bounds__(256)
void mlp1_kernel(const u16* __restrict__ xb,
                 const float* __restrict__ wgB, const float* __restrict__ wuB,
                 u16* __restrict__ actOut,
                 const int* __restrict__ counts, const int* __restrict__ offsets,
                 const int* __restrict__ tokIdx,
                 int Kdim, int Ndim) {
  const int e = blockIdx.z;
  int cnt, rowOff; const int* tl;
  if (counts) { cnt = counts[e]; rowOff = offsets[e]; tl = tokIdx + e * T_TOK; }
  else        { cnt = T_TOK;     rowOff = 0;          tl = nullptr; }
  const int padded = (cnt + 15) & ~15;
  const int mBase = blockIdx.y * 128;
  if (mBase >= padded) return;
  const int n0 = blockIdx.x * 64;
  const float* __restrict__ wg = wgB + (size_t)e * Kdim * Ndim;
  const float* __restrict__ wu = wuB + (size_t)e * Kdim * Ndim;

  __shared__ alignas(16) u16 lA[128 * 40];   // activations, [m][k] bf16
  __shared__ alignas(16) u16 lBg[64 * 40];   // gate weights, transposed [n][k] bf16
  __shared__ alignas(16) u16 lBu[64 * 40];   // up weights,   transposed [n][k] bf16
  __shared__ int lTok[128];

  const int tid = threadIdx.x;
  if (tid < 128) {
    int r = mBase + tid;
    lTok[tid] = tl ? tl[(r < cnt) ? r : 0] : r;
  }
  __syncthreads();

  const int wave = tid >> 5, lane = tid & 31;
  const int sel = lane >> 4, nlo = lane & 15;
  const int rowA = tid >> 1, c2 = tid & 1;
  const u16* __restrict__ srcArow = xb + (size_t)lTok[rowA] * H_DIM;
  const u32 aDst = lds_off(&lA[rowA * 40 + c2 * 16]);

  // B staging plan: items = mat(2) x kpair(16) x n4(16); 2 items/thread
  const int kpB = (tid >> 4) & 15;
  const int n4B = tid & 3;        // unused; real decode below

  v8f accG[4] = {}; v8f accU[4] = {};
  const int ar = wave * 16 + nlo;

  for (int k0 = 0; k0 < Kdim; k0 += 32) {
    // A tile: async DMA global -> LDS (2 x 16B per thread)
    const u16* ga = srcArow + k0 + c2 * 16;
    async_b128(aDst, ga);
    async_b128(aDst + 16, ga + 8);
    // B tiles: fp32 -> bf16, k-pairs packed into u32, transposed [n][k]
#pragma unroll
    for (int it = 0; it < 2; ++it) {
      const int i   = tid + it * 256;
      const int mat = i >> 8;           // 0=gate 1=up (uniform per it)
      const int kp  = (i >> 4) & 15;
      const int n4  = i & 15;
      const float* w0 = (mat ? wu : wg) + (size_t)(k0 + 2 * kp) * Ndim + n0 + n4 * 4;
      float4 f0 = *(const float4*)w0;
      float4 f1 = *(const float4*)(w0 + Ndim);
      u16* dst = mat ? lBu : lBg;
      const int nb = n4 * 4;
      *(u32*)&dst[(nb + 0) * 40 + kp * 2] = (u32)f2bf(f0.x) | ((u32)f2bf(f1.x) << 16);
      *(u32*)&dst[(nb + 1) * 40 + kp * 2] = (u32)f2bf(f0.y) | ((u32)f2bf(f1.y) << 16);
      *(u32*)&dst[(nb + 2) * 40 + kp * 2] = (u32)f2bf(f0.z) | ((u32)f2bf(f1.z) << 16);
      *(u32*)&dst[(nb + 3) * 40 + kp * 2] = (u32)f2bf(f0.w) | ((u32)f2bf(f1.w) << 16);
      if (k0 + 32 < Kdim) __builtin_prefetch(w0 + 32 * (size_t)Ndim, 0, 0);
    }
    wait_async0();
    __syncthreads();

    Frag fa;
    fa.q[0] = *(const uint4*)&lA[ar * 40 + sel * 8];
    fa.q[1] = *(const uint4*)&lA[ar * 40 + 16 + sel * 8];
#pragma unroll
    for (int ns = 0; ns < 4; ++ns) {
      const int nl = ns * 16 + nlo;
      Frag fg, fu;
      fg.q[0] = *(const uint4*)&lBg[nl * 40 + sel * 8];
      fg.q[1] = *(const uint4*)&lBg[nl * 40 + 16 + sel * 8];
      fu.q[0] = *(const uint4*)&lBu[nl * 40 + sel * 8];
      fu.q[1] = *(const uint4*)&lBu[nl * 40 + 16 + sel * 8];
      accG[ns] = __builtin_amdgcn_wmma_f32_16x16x32_bf16(false, fa.v, false, fg.v, (short)0, accG[ns], false, false);
      accU[ns] = __builtin_amdgcn_wmma_f32_16x16x32_bf16(false, fa.v, false, fu.v, (short)0, accU[ns], false, false);
    }
    __syncthreads();
  }

  const int mloc = wave * 16 + sel * 8;
#pragma unroll
  for (int ns = 0; ns < 4; ++ns) {
#pragma unroll
    for (int r = 0; r < 8; ++r) {
      int m = mBase + mloc + r;
      if (m < cnt) {
        float a = silu_f(accG[ns][r]) * accU[ns][r];
        actOut[(size_t)(rowOff + m) * Ndim + n0 + ns * 16 + nlo] = f2bf(a);
      }
    }
  }
  (void)kpB; (void)n4B;
}

// ---------------- 5/6) down-projection GEMM (shared writes, routed atomically combines) ----------------
__global__ __launch_bounds__(256)
void mlp2_kernel(const u16* __restrict__ act,
                 const float* __restrict__ wdB,
                 float* __restrict__ out,
                 const int* __restrict__ counts, const int* __restrict__ offsets,
                 const int* __restrict__ tokIdx, const float* __restrict__ combine,
                 int Kdim) {
  const int e = blockIdx.z;
  int cnt, rowOff; const int* tl;
  if (counts) { cnt = counts[e]; rowOff = offsets[e]; tl = tokIdx + e * T_TOK; }
  else        { cnt = T_TOK;     rowOff = 0;          tl = nullptr; }
  const int padded = (cnt + 15) & ~15;
  const int mBase = blockIdx.y * 128;
  if (mBase >= padded) return;
  const int n0 = blockIdx.x * 64;
  const float* __restrict__ wd = wdB + (size_t)e * Kdim * H_DIM;

  __shared__ alignas(16) u16 lA[128 * 40];
  __shared__ alignas(16) u16 lB[64 * 40];
  __shared__ int lTok[128];

  const int tid = threadIdx.x;
  if (tid < 128) {
    int r = mBase + tid;
    lTok[tid] = tl ? ((r < cnt) ? tl[r] : 0) : r;
  }
  __syncthreads();

  const int wave = tid >> 5, lane = tid & 31;
  const int sel = lane >> 4, nlo = lane & 15;
  const int rowA = tid >> 1, c2 = tid & 1;
  const u16* __restrict__ srcArow = act + (size_t)(rowOff + mBase + rowA) * Kdim;
  const u32 aDst = lds_off(&lA[rowA * 40 + c2 * 16]);

  // B staging: items = kpair(16) x n4(16) = 256 -> 1 item/thread
  const int kp = (tid >> 4) & 15;
  const int n4 = tid & 15;
  const float* __restrict__ wrow = wd + (size_t)(2 * kp) * H_DIM + n0 + n4 * 4;

  v8f acc[4] = {};
  const int ar = wave * 16 + nlo;

  for (int k0 = 0; k0 < Kdim; k0 += 32) {
    const u16* ga = srcArow + k0 + c2 * 16;
    async_b128(aDst, ga);
    async_b128(aDst + 16, ga + 8);

    const float* w0 = wrow + (size_t)k0 * H_DIM;
    float4 f0 = *(const float4*)w0;
    float4 f1 = *(const float4*)(w0 + H_DIM);
    const int nb = n4 * 4;
    *(u32*)&lB[(nb + 0) * 40 + kp * 2] = (u32)f2bf(f0.x) | ((u32)f2bf(f1.x) << 16);
    *(u32*)&lB[(nb + 1) * 40 + kp * 2] = (u32)f2bf(f0.y) | ((u32)f2bf(f1.y) << 16);
    *(u32*)&lB[(nb + 2) * 40 + kp * 2] = (u32)f2bf(f0.z) | ((u32)f2bf(f1.z) << 16);
    *(u32*)&lB[(nb + 3) * 40 + kp * 2] = (u32)f2bf(f0.w) | ((u32)f2bf(f1.w) << 16);
    if (k0 + 32 < Kdim) __builtin_prefetch(w0 + 32 * (size_t)H_DIM, 0, 0);

    wait_async0();
    __syncthreads();

    Frag fa;
    fa.q[0] = *(const uint4*)&lA[ar * 40 + sel * 8];
    fa.q[1] = *(const uint4*)&lA[ar * 40 + 16 + sel * 8];
#pragma unroll
    for (int ns = 0; ns < 4; ++ns) {
      const int nl = ns * 16 + nlo;
      Frag fb;
      fb.q[0] = *(const uint4*)&lB[nl * 40 + sel * 8];
      fb.q[1] = *(const uint4*)&lB[nl * 40 + 16 + sel * 8];
      acc[ns] = __builtin_amdgcn_wmma_f32_16x16x32_bf16(false, fa.v, false, fb.v, (short)0, acc[ns], false, false);
    }
    __syncthreads();
  }

  const int mloc = wave * 16 + sel * 8;
#pragma unroll
  for (int ns = 0; ns < 4; ++ns) {
#pragma unroll
    for (int r = 0; r < 8; ++r) {
      int m = mBase + mloc + r;
      if (m < cnt) {
        int tok = lTok[mloc + r];
        float v = acc[ns][r];
        float* dst = out + (size_t)tok * H_DIM + n0 + ns * 16 + nlo;
        if (combine) {
          v *= combine[(size_t)tok * E_NUM + e];
          unsafeAtomicAdd(dst, v);   // global_atomic_add_f32, no return
        } else {
          *dst = v;
        }
      }
    }
  }
}

// ---------------- launch ----------------
extern "C" void kernel_launch(void* const* d_in, const int* in_sizes, int n_in,
                              void* d_out, int out_size, void* d_ws, size_t ws_size,
                              hipStream_t stream) {
  (void)in_sizes; (void)n_in; (void)out_size; (void)ws_size;
  const float* x         = (const float*)d_in[0];
  const float* gate_w    = (const float*)d_in[1];
  const float* corr_bias = (const float*)d_in[2];
  const float* w_gate    = (const float*)d_in[3];
  const float* w_up      = (const float*)d_in[4];
  const float* w_down    = (const float*)d_in[5];
  const float* sw_gate   = (const float*)d_in[6];
  const float* sw_up     = (const float*)d_in[7];
  const float* sw_down   = (const float*)d_in[8];
  float* out = (float*)d_out;

  char* p = (char*)d_ws;
  auto carve = [&](size_t bytes) -> void* {
    void* r = (void*)p;
    p += (bytes + 255) & ~(size_t)255;
    return r;
  };
  u16*   xb      = (u16*)  carve((size_t)T_TOK * H_DIM * 2);
  float* combine = (float*)carve((size_t)T_TOK * E_NUM * 4);
  int*   counts  = (int*)  carve(E_NUM * 4);
  int*   offsets = (int*)  carve((E_NUM + 1) * 4);
  int*   tokIdx  = (int*)  carve((size_t)E_NUM * T_TOK * 4);
  u16*   S1      = (u16*)  carve((size_t)T_TOK * IS_DIM * 2);
  u16*   Aslab   = (u16*)  carve(((size_t)T_TOK * 8 + 1024) * I_DIM * 2);

  hipMemsetAsync(counts, 0, E_NUM * 4, stream);
  cast_bf16_kernel<<<(T_TOK * H_DIM / 4 + 255) / 256, 256, 0, stream>>>(x, xb, T_TOK * H_DIM / 4);
  route_kernel<<<T_TOK, 32, 0, stream>>>(x, gate_w, corr_bias, combine, counts, tokIdx);
  scan_kernel<<<1, 32, 0, stream>>>(counts, offsets);

  // shared experts: S1 = silu(x@sw_gate) * (x@sw_up)
  mlp1_kernel<<<dim3(IS_DIM / 64, 8, 1), 256, 0, stream>>>(
      xb, sw_gate, sw_up, S1, nullptr, nullptr, nullptr, H_DIM, IS_DIM);
  // routed experts: Aslab[rows(e), I] = silu(x_g@Wg[e]) * (x_g@Wu[e])
  mlp1_kernel<<<dim3(I_DIM / 64, 8, E_NUM), 256, 0, stream>>>(
      xb, w_gate, w_up, Aslab, counts, offsets, tokIdx, H_DIM, I_DIM);
  // out = S1 @ sw_down   (base write)
  mlp2_kernel<<<dim3(H_DIM / 64, 8, 1), 256, 0, stream>>>(
      S1, sw_down, out, nullptr, nullptr, nullptr, nullptr, IS_DIM);
  // out += combine[t,e] * (Aslab @ w_down[e])   (atomic f32 add)
  mlp2_kernel<<<dim3(H_DIM / 64, 8, E_NUM), 256, 0, stream>>>(
      Aslab, w_down, out, counts, offsets, tokIdx, combine, I_DIM);
}
